// MyGptLayer_36850819400221
// MI455X (gfx1250) — matrix operand order repaired and checked by reading, not compiled, not run on hardware
//
#include <hip/hip_runtime.h>
#include <hip/hip_bf16.h>

typedef __attribute__((ext_vector_type(16))) __bf16 v16bf;
typedef __attribute__((ext_vector_type(8)))  __bf16 v8bf;
typedef __attribute__((ext_vector_type(8)))  float  v8f;

constexpr int E_ = 2048;   // embed dim
constexpr int S_ = 2048;   // seq len
constexpr int B_ = 2;      // batch
constexpr int H_ = 16;     // heads
constexpr int D_ = 128;    // head dim
constexpr int M_ = S_ * B_;  // 4096 rows of the [S*B, E] activation matrix

__device__ __forceinline__ unsigned short f32_to_bf16(float f) {
  unsigned int u = __builtin_bit_cast(unsigned int, f);
  unsigned int r = u + 0x7fffu + ((u >> 16) & 1u);   // round-to-nearest-even
  return (unsigned short)(r >> 16);
}

__device__ __forceinline__ v16bf combine16(v8bf lo, v8bf hi) {
  return __builtin_shufflevector(lo, hi, 0,1,2,3,4,5,6,7,8,9,10,11,12,13,14,15);
}

// ---- cross-lane reductions within 16-lane groups, VALU-only (no DS port) ----
__device__ __forceinline__ float xor1_dpp(float x) {   // quad_perm [1,0,3,2]
  int i = __builtin_bit_cast(int, x);
  int r = __builtin_amdgcn_update_dpp(i, i, 0xB1, 0xf, 0xf, true);
  return __builtin_bit_cast(float, r);
}
__device__ __forceinline__ float xor2_dpp(float x) {   // quad_perm [2,3,0,1]
  int i = __builtin_bit_cast(int, x);
  int r = __builtin_amdgcn_update_dpp(i, i, 0x4E, 0xf, 0xf, true);
  return __builtin_bit_cast(float, r);
}
__device__ __forceinline__ float xor4_pl(float x) {    // lane -> lane^4 in row16
  int i = __builtin_bit_cast(int, x);
  int r = __builtin_amdgcn_permlane16(i, i, 0x32107654u, 0xBA98FEDCu, false, false);
  return __builtin_bit_cast(float, r);
}
__device__ __forceinline__ float xor8_pl(float x) {    // lane -> lane^8 in row16
  int i = __builtin_bit_cast(int, x);
  int r = __builtin_amdgcn_permlane16(i, i, 0xFEDCBA98u, 0x76543210u, false, false);
  return __builtin_bit_cast(float, r);
}
__device__ __forceinline__ float redmax16(float v) {
  v = fmaxf(v, xor1_dpp(v));
  v = fmaxf(v, xor2_dpp(v));
  v = fmaxf(v, xor4_pl(v));
  v = fmaxf(v, xor8_pl(v));
  return v;
}
__device__ __forceinline__ float redsum16(float v) {
  v += xor1_dpp(v);
  v += xor2_dpp(v);
  v += xor4_pl(v);
  v += xor8_pl(v);
  return v;
}

// ---- CDNA5 async global->LDS copy (ASYNCcnt-tracked), 16B per lane ----------
__device__ __forceinline__ unsigned lds_off(const void* p) {
  // generic pointer to LDS: low 32 bits are the wave-relative LDS byte offset
  return (unsigned)(unsigned long long)p;
}
__device__ __forceinline__ void async_b128(unsigned dst_lds, const void* src) {
  unsigned long long ga = (unsigned long long)src;
  asm volatile("global_load_async_to_lds_b128 %0, %1, off"
               :: "v"(dst_lds), "v"(ga) : "memory");
}
__device__ __forceinline__ void wait_async0() {
  asm volatile("s_wait_asynccnt 0" ::: "memory");
}

// ---------------------------------------------------------------------------
// Transposing f32 -> bf16 weight conversion: out[n*R + k] = in[k*C + n]
// ---------------------------------------------------------------------------
__global__ __launch_bounds__(256) void transpose_f32_to_bf16(
    const float* __restrict__ in, unsigned short* __restrict__ out, int R, int C)
{
  __shared__ float tile[32][33];
  const int tx = threadIdx.x & 31, ty = threadIdx.x >> 5;
  const int r0 = blockIdx.y * 32, c0 = blockIdx.x * 32;
  #pragma unroll
  for (int i = 0; i < 32; i += 8)
    tile[ty + i][tx] = in[(size_t)(r0 + ty + i) * C + (c0 + tx)];
  __syncthreads();
  #pragma unroll
  for (int i = 0; i < 32; i += 8)
    out[(size_t)(c0 + ty + i) * R + (r0 + tx)] = f32_to_bf16(tile[tx][ty + i]);
}

__global__ __launch_bounds__(256) void convert_f32_to_bf16(
    const float* __restrict__ in, unsigned short* __restrict__ out, int n)
{
  int i = blockIdx.x * 256 + threadIdx.x;
  if (i < n) out[i] = f32_to_bf16(in[i]);
}

// ---------------------------------------------------------------------------
// bf16 GEMM: C[M,N] = A[M,K] * W[K,N] + bias ;  W given transposed Wt[N,K].
// 128x128 block tile, BK=32, double-buffered async global->LDS pipeline.
// 8 waves, each wave computes 64x32 (4x2 wmma tiles).
// MODE: 0 = bf16 row-major out
//       1 = q: scale by 1/sqrt(d), bf16 scatter to [B,H,S,D]
//       2 = k: bf16 scatter to [B,H,S,D] + f32 scatter to present[0]
//       3 = v: bf16 scatter to [B,H,D,S] (d-major) + f32 scatter to present[1]
//       4 = f32 row-major out
// ---------------------------------------------------------------------------
template<int MODE>
__global__ __launch_bounds__(256) void gemm_bf16(
    const unsigned short* __restrict__ A,
    const unsigned short* __restrict__ Bt,
    const float* __restrict__ bias,
    unsigned short* __restrict__ outB,
    float* __restrict__ outF)
{
  constexpr int K = E_;
  constexpr int N = E_;
  constexpr int LDT = 40;                       // padded LDS row stride (elems)
  __shared__ unsigned short Asm[2][128 * LDT];
  __shared__ unsigned short Bsm[2][128 * LDT];

  const int tid  = threadIdx.x;
  const int wid  = tid >> 5;
  const int lane = tid & 31;
  const int wy   = wid >> 2;                    // 0..1 -> 64-row slab
  const int wx   = wid & 3;                     // 0..3 -> 32-col slab
  const int bm0  = blockIdx.y * 128;
  const int bn0  = blockIdx.x * 128;
  const int lrow = tid >> 1;                    // 0..127 tile row to load
  const int lcol = (tid & 1) * 16;              // 0 / 16 elem offset
  const int nlane = lane & 15;
  const int hlf   = lane >> 4;

  v8f acc[4][2] = {};

  auto issue = [&](int buf, int k0) {
    const unsigned short* ga = A  + (size_t)(bm0 + lrow) * K + k0 + lcol;
    const unsigned short* gb = Bt + (size_t)(bn0 + lrow) * K + k0 + lcol;
    unsigned la = lds_off(&Asm[buf][lrow * LDT + lcol]);
    unsigned lb = lds_off(&Bsm[buf][lrow * LDT + lcol]);
    async_b128(la,      ga);
    async_b128(la + 16, ga + 8);
    async_b128(lb,      gb);
    async_b128(lb + 16, gb + 8);
  };

  issue(0, 0);
  for (int k0 = 0; k0 < K; k0 += 32) {
    const int buf = (k0 >> 5) & 1;
    wait_async0();                 // my async copies into buf are done
    __syncthreads();               // everyone's are done; prev reads finished
    if (k0 + 32 < K) issue(buf ^ 1, k0 + 32);   // prefetch next tile

    // A fragment: lane (m = lane&15, half = lane>>4) holds K = {8h..8h+7, 16+8h..16+8h+7}
    v16bf af[4];
    #pragma unroll
    for (int mf = 0; mf < 4; ++mf) {
      const unsigned short* p = &Asm[buf][(wy * 64 + mf * 16 + nlane) * LDT];
      af[mf] = combine16(*(const v8bf*)(p + 8 * hlf),
                         *(const v8bf*)(p + 16 + 8 * hlf));
    }
    // B fragment: lane (n = lane&15, half = lane>>4) holds K = {8h..8h+7, 16+8h..16+8h+7}
    v16bf bfv[2];
    #pragma unroll
    for (int nf = 0; nf < 2; ++nf) {
      const unsigned short* p =
          &Bsm[buf][(wx * 32 + nf * 16 + nlane) * LDT];
      bfv[nf] = combine16(*(const v8bf*)(p + 8 * hlf), *(const v8bf*)(p + 16 + 8 * hlf));
    }
    #pragma unroll
    for (int mf = 0; mf < 4; ++mf)
      #pragma unroll
      for (int nf = 0; nf < 2; ++nf)
        acc[mf][nf] = __builtin_amdgcn_wmma_f32_16x16x32_bf16(
            false, af[mf], false, bfv[nf], (short)0, acc[mf][nf], false, false);
  }

  // Epilogue. C layout: lane n = lane&15, row m = r + 8*(lane>>4), r = VGPR index.
  #pragma unroll
  for (int nf = 0; nf < 2; ++nf) {
    const int gcol = bn0 + wx * 32 + nf * 16 + nlane;
    const float bv = bias[gcol];
    #pragma unroll
    for (int mf = 0; mf < 4; ++mf) {
      #pragma unroll
      for (int r = 0; r < 8; ++r) {
        const int grow = bm0 + wy * 64 + mf * 16 + r + 8 * hlf;
        float v = acc[mf][nf][r] + bv;
        if (MODE == 0) {
          outB[(size_t)grow * N + gcol] = f32_to_bf16(v);
        } else if (MODE == 4) {
          outF[(size_t)grow * N + gcol] = v;
        } else {
          const int s = grow >> 1, b = grow & 1;        // row = s*B + b, B=2
          const int h = gcol >> 7, dd = gcol & 127;     // col = h*128 + dd
          if (MODE == 1) {
            outB[((size_t)(b * H_ + h) * S_ + s) * D_ + dd] =
                f32_to_bf16(v * 0.08838834764831845f);  // 1/sqrt(128)
          } else if (MODE == 2) {
            outB[((size_t)(b * H_ + h) * S_ + s) * D_ + dd] = f32_to_bf16(v);
            outF[((size_t)(b * S_ + s) * H_ + h) * D_ + dd] = v;
          } else { // MODE == 3
            outB[((size_t)(b * H_ + h) * D_ + dd) * S_ + s] = f32_to_bf16(v);
            outF[((size_t)(b * S_ + s) * H_ + h) * D_ + dd] = v;
          }
        }
      }
    }
  }
}

// ---------------------------------------------------------------------------
// Flash attention: grid (S/128, B*H), 8 waves; each wave owns 16 query rows.
// qh/kh: bf16 [B*H, S, D];  vt: bf16 [B*H, D, S];  ctx: bf16 [S*B, E].
// Double-buffered async K/V tiles of 32 keys.
// ---------------------------------------------------------------------------
__global__ __launch_bounds__(256) void attn_flash(
    const unsigned short* __restrict__ qh,
    const unsigned short* __restrict__ kh,
    const unsigned short* __restrict__ vt,
    unsigned short* __restrict__ ctx)
{
  constexpr int LK = 136;                       // 128 + pad
  constexpr int LV = 40;                        // 32 + pad
  __shared__ unsigned short ksm[2][32 * LK];    // 32 keys x 128 d
  __shared__ unsigned short vsm[2][128 * LV];   // 128 d  x 32 keys
  __shared__ unsigned short psm[8 * 16 * 32];   // per-wave P tile (16x32)

  const int tid = threadIdx.x, wid = tid >> 5, lane = tid & 31;
  const int nlane = lane & 15, hlf = lane >> 4;
  const int bh = blockIdx.y;
  const int b = bh >> 4, h = bh & 15;           // bh = b*H + h
  const int q0 = blockIdx.x * 128 + wid * 16;
  const size_t base = (size_t)bh * S_ * D_;

  // Q fragments (16 rows x 128 d as 4 chunks of K=32)
  v16bf qf[4];
  #pragma unroll
  for (int c = 0; c < 4; ++c) {
    const unsigned short* p = qh + base + (size_t)(q0 + nlane) * D_ + c * 32;
    qf[c] = combine16(*(const v8bf*)(p + 8 * hlf),
                      *(const v8bf*)(p + 16 + 8 * hlf));
  }

  float mrow[8], lrow[8], alpha[8];
  v8f o[8] = {};
  #pragma unroll
  for (int r = 0; r < 8; ++r) { mrow[r] = -3.0e38f; lrow[r] = 0.f; }

  const int krow = tid >> 3, kch = (tid & 7) * 16;
  const int vrow = tid >> 1, vch = (tid & 1) * 16;
  unsigned short* pw = psm + wid * 512;

  auto issue = [&](int buf, int t0) {
    const unsigned short* gk = kh + base + (size_t)(t0 + krow) * D_ + kch;
    unsigned lk = lds_off(&ksm[buf][krow * LK + kch]);
    async_b128(lk,      gk);
    async_b128(lk + 16, gk + 8);
    const unsigned short* gv =
        vt + (size_t)bh * D_ * S_ + (size_t)vrow * S_ + t0 + vch;
    unsigned lv = lds_off(&vsm[buf][vrow * LV + vch]);
    async_b128(lv,      gv);
    async_b128(lv + 16, gv + 8);
  };

  issue(0, 0);
  for (int t0 = 0; t0 < S_; t0 += 32) {
    const int buf = (t0 >> 5) & 1;
    wait_async0();
    __syncthreads();
    if (t0 + 32 < S_) issue(buf ^ 1, t0 + 32);

    // scores: q(16x128) . k^T(128x32) -> two 16x16 tiles
    v8f sc0 = {}, sc1 = {};
    #pragma unroll
    for (int c = 0; c < 4; ++c) {
      const unsigned short* p0 = &ksm[buf][nlane * LK + c * 32];
      v16bf kf0 = combine16(*(const v8bf*)(p0 + 8 * hlf), *(const v8bf*)(p0 + 16 + 8 * hlf));
      sc0 = __builtin_amdgcn_wmma_f32_16x16x32_bf16(false, qf[c], false, kf0,
                                                    (short)0, sc0, false, false);
      const unsigned short* p1 = &ksm[buf][(16 + nlane) * LK + c * 32];
      v16bf kf1 = combine16(*(const v8bf*)(p1 + 8 * hlf), *(const v8bf*)(p1 + 16 + 8 * hlf));
      sc1 = __builtin_amdgcn_wmma_f32_16x16x32_bf16(false, qf[c], false, kf1,
                                                    (short)0, sc1, false, false);
    }

    // online softmax (rows live across 16-lane groups per C layout)
    #pragma unroll
    for (int r = 0; r < 8; ++r) {
      float mx = redmax16(fmaxf(sc0[r], sc1[r]));
      float mnew = fmaxf(mrow[r], mx);
      float a = __expf(mrow[r] - mnew);
      float p0 = __expf(sc0[r] - mnew);
      float p1 = __expf(sc1[r] - mnew);
      sc0[r] = p0; sc1[r] = p1;
      lrow[r] = lrow[r] * a + redsum16(p0 + p1);
      mrow[r] = mnew;
      alpha[r] = a;
    }
    #pragma unroll
    for (int f = 0; f < 8; ++f)
      #pragma unroll
      for (int r = 0; r < 8; ++r)
        o[f][r] *= alpha[r];

    // P: C-fragment layout -> LDS -> A-fragment layout (wave-local round trip)
    #pragma unroll
    for (int r = 0; r < 8; ++r) {
      pw[(r + 8 * hlf) * 32 + nlane]      = f32_to_bf16(sc0[r]);
      pw[(r + 8 * hlf) * 32 + 16 + nlane] = f32_to_bf16(sc1[r]);
    }
    __builtin_amdgcn_wave_barrier();
    asm volatile("" ::: "memory");
    const unsigned short* pp = pw + nlane * 32;
    v16bf pf = combine16(*(const v8bf*)(pp + 8 * hlf),
                         *(const v8bf*)(pp + 16 + 8 * hlf));

    // O(16x128) += P(16x32) . V(32x128), 8 n-chunks
    #pragma unroll
    for (int f = 0; f < 8; ++f) {
      const unsigned short* pv = &vsm[buf][(f * 16 + nlane) * LV];
      v16bf vf = combine16(*(const v8bf*)(pv + 8 * hlf), *(const v8bf*)(pv + 16 + 8 * hlf));
      o[f] = __builtin_amdgcn_wmma_f32_16x16x32_bf16(false, pf, false, vf,
                                                     (short)0, o[f], false, false);
    }
  }

  float inv[8];
  #pragma unroll
  for (int r = 0; r < 8; ++r) inv[r] = 1.f / lrow[r];
  #pragma unroll
  for (int f = 0; f < 8; ++f) {
    const int dcol = f * 16 + nlane;
    #pragma unroll
    for (int r = 0; r < 8; ++r) {
      const int srow = q0 + r + 8 * hlf;
      ctx[(size_t)(srow * B_ + b) * E_ + h * D_ + dcol] =
          f32_to_bf16(o[f][r] * inv[r]);
    }
  }
}

// ---------------------------------------------------------------------------
extern "C" void kernel_launch(void* const* d_in, const int* in_sizes, int n_in,
                              void* d_out, int out_size, void* d_ws, size_t ws_size,
                              hipStream_t stream)
{
  (void)in_sizes; (void)n_in; (void)out_size; (void)ws_size;
  const float* x = (const float*)d_in[0];
  const float* W[6]    = {(const float*)d_in[1], (const float*)d_in[3],
                          (const float*)d_in[5], (const float*)d_in[7],
                          (const float*)d_in[9], (const float*)d_in[11]};
  const float* bias[6] = {(const float*)d_in[2], (const float*)d_in[4],
                          (const float*)d_in[6], (const float*)d_in[8],
                          (const float*)d_in[10], (const float*)d_in[12]};
  float* out = (float*)d_out;
  float* present0 = out + (size_t)M_ * E_;
  float* present1 = present0 + (size_t)B_ * S_ * H_ * D_;

  char* ws = (char*)d_ws;
  const size_t WSZ = (size_t)E_ * E_ * sizeof(unsigned short);  // 8 MB
  const size_t ASZ = (size_t)M_ * E_ * sizeof(unsigned short);  // 16 MB
  unsigned short* Wt[6];
  for (int i = 0; i < 6; ++i) Wt[i] = (unsigned short*)(ws + (size_t)i * WSZ);
  unsigned short* xb  = (unsigned short*)(ws + 6 * WSZ);            // x bf16 / ctx
  unsigned short* qhb = (unsigned short*)(ws + 6 * WSZ + ASZ);      // q / attn_out
  unsigned short* khb = (unsigned short*)(ws + 6 * WSZ + 2 * ASZ);  // k / ffn mid
  unsigned short* vtb = (unsigned short*)(ws + 6 * WSZ + 3 * ASZ);  // v (d-major)

  dim3 tg(E_ / 32, E_ / 32);
  for (int i = 0; i < 6; ++i)
    transpose_f32_to_bf16<<<tg, 256, 0, stream>>>(W[i], Wt[i], E_, E_);
  convert_f32_to_bf16<<<(M_ * E_ + 255) / 256, 256, 0, stream>>>(x, xb, M_ * E_);

  dim3 gg(E_ / 128, M_ / 128);
  gemm_bf16<1><<<gg, 256, 0, stream>>>(xb, Wt[0], bias[0], qhb, nullptr);
  gemm_bf16<2><<<gg, 256, 0, stream>>>(xb, Wt[1], bias[1], khb, present0);
  gemm_bf16<3><<<gg, 256, 0, stream>>>(xb, Wt[2], bias[2], vtb, present1);

  attn_flash<<<dim3(S_ / 128, B_ * H_), 256, 0, stream>>>(qhb, khb, vtb, xb);

  gemm_bf16<0><<<gg, 256, 0, stream>>>(xb,  Wt[3], bias[3], qhb, nullptr);  // Wo
  gemm_bf16<0><<<gg, 256, 0, stream>>>(qhb, Wt[4], bias[4], khb, nullptr);  // W1
  gemm_bf16<4><<<gg, 256, 0, stream>>>(khb, Wt[5], bias[5], nullptr, out);  // W2
}
